// Decoder_5179730559388
// MI455X (gfx1250) — matrix operand order repaired and checked by reading, not compiled, not run on hardware
//
#include <hip/hip_runtime.h>

typedef __bf16 bf16_t;
typedef __attribute__((ext_vector_type(16))) __bf16 v16bf;
typedef __attribute__((ext_vector_type(8)))  __bf16 v8bf;
typedef __attribute__((ext_vector_type(8)))  float  v8f;

__device__ inline bf16_t f2bf(float f) {
    unsigned u = __float_as_uint(f);
    u += 0x7FFFu + ((u >> 16) & 1u);               // round-to-nearest-even
    unsigned short h = (unsigned short)(u >> 16);
    return __builtin_bit_cast(bf16_t, h);
}

// ---------------------------------------------------------------- converts
__global__ void k_f32_to_bf16(const float* __restrict__ src,
                              bf16_t* __restrict__ dst, long n) {
    long i4 = ((long)blockIdx.x * blockDim.x + threadIdx.x) * 4;
    if (i4 + 3 >= n) {
        for (long i = i4; i < n; ++i) dst[i] = f2bf(src[i]);
        return;
    }
    float4 v = *(const float4*)(src + i4);
    union { bf16_t b[4]; unsigned long long u; } o;
    o.b[0] = f2bf(v.x); o.b[1] = f2bf(v.y); o.b[2] = f2bf(v.z); o.b[3] = f2bf(v.w);
    *(unsigned long long*)(dst + i4) = o.u;
}

// embedded = target + pos_emb  (f32 + bf16 copies)
__global__ void k_embed(const float* __restrict__ tgt, const float* __restrict__ pos,
                        float* __restrict__ outF, bf16_t* __restrict__ outB,
                        long TE, long n) {
    long i4 = ((long)blockIdx.x * blockDim.x + threadIdx.x) * 4;
    if (i4 >= n) return;
    float4 a = *(const float4*)(tgt + i4);
    float4 p = *(const float4*)(pos + (i4 % TE));
    float4 r = make_float4(a.x + p.x, a.y + p.y, a.z + p.z, a.w + p.w);
    *(float4*)(outF + i4) = r;
    union { bf16_t b[4]; unsigned long long u; } o;
    o.b[0] = f2bf(r.x); o.b[1] = f2bf(r.y); o.b[2] = f2bf(r.z); o.b[3] = f2bf(r.w);
    *(unsigned long long*)(outB + i4) = o.u;
}

// im2col for causal K=3 conv: row (b,t) = [x[t-2] | x[t-1] | x[t]], zero-padded
__global__ void k_im2col(const float* __restrict__ x, bf16_t* __restrict__ col,
                         int T, int H, long n) {
    long i4 = ((long)blockIdx.x * blockDim.x + threadIdx.x) * 4;
    if (i4 >= n) return;
    long rowc = 3L * H;
    long m = i4 / rowc;
    int  c = (int)(i4 - m * rowc);
    int  k = c / H, h = c - k * H;
    int  b = (int)(m / T), t = (int)(m - (long)b * T);
    int  ts = t - 2 + k;
    float4 v = make_float4(0.f, 0.f, 0.f, 0.f);
    if (ts >= 0) v = *(const float4*)(x + ((long)(b * T + ts)) * H + h);
    union { bf16_t b[4]; unsigned long long u; } o;
    o.b[0] = f2bf(v.x); o.b[1] = f2bf(v.y); o.b[2] = f2bf(v.z); o.b[3] = f2bf(v.w);
    *(unsigned long long*)(col + i4) = o.u;
}

// GLU: out = a * sigmoid(g)   in2h is [M, 2H]
__global__ void k_glu(const float* __restrict__ in2h,
                      float* __restrict__ outF, bf16_t* __restrict__ outB,
                      int H, long n) {
    long i4 = ((long)blockIdx.x * blockDim.x + threadIdx.x) * 4;
    if (i4 >= n) return;
    long m = i4 / H;
    int  h = (int)(i4 - m * H);
    const float* ap = in2h + m * 2L * H + h;
    float4 a = *(const float4*)ap;
    float4 g = *(const float4*)(ap + H);
    float4 r;
    r.x = a.x * (1.f / (1.f + __expf(-g.x)));
    r.y = a.y * (1.f / (1.f + __expf(-g.y)));
    r.z = a.z * (1.f / (1.f + __expf(-g.z)));
    r.w = a.w * (1.f / (1.f + __expf(-g.w)));
    *(float4*)(outF + i4) = r;
    union { bf16_t b[4]; unsigned long long u; } o;
    o.b[0] = f2bf(r.x); o.b[1] = f2bf(r.y); o.b[2] = f2bf(r.z); o.b[3] = f2bf(r.w);
    *(unsigned long long*)(outB + i4) = o.u;
}

// row softmax over S=1024, 256 threads/row, 4 elems/thread
__global__ void k_softmax(const float* __restrict__ e,
                          float* __restrict__ outF, bf16_t* __restrict__ outB, int S) {
    __shared__ float red[256];
    int row = blockIdx.x, tid = threadIdx.x;
    const float* er = e + (long)row * S + tid * 4;
    float4 v = *(const float4*)er;
    float mx = fmaxf(fmaxf(v.x, v.y), fmaxf(v.z, v.w));
    red[tid] = mx; __syncthreads();
    for (int s = 128; s > 0; s >>= 1) {
        if (tid < s) red[tid] = fmaxf(red[tid], red[tid + s]);
        __syncthreads();
    }
    float rmax = red[0]; __syncthreads();
    float4 x;
    x.x = __expf(v.x - rmax); x.y = __expf(v.y - rmax);
    x.z = __expf(v.z - rmax); x.w = __expf(v.w - rmax);
    red[tid] = x.x + x.y + x.z + x.w; __syncthreads();
    for (int s = 128; s > 0; s >>= 1) {
        if (tid < s) red[tid] += red[tid + s];
        __syncthreads();
    }
    float inv = 1.f / red[0];
    x.x *= inv; x.y *= inv; x.z *= inv; x.w *= inv;
    long o = (long)row * S + tid * 4;
    *(float4*)(outF + o) = x;
    union { bf16_t b[4]; unsigned long long u; } ob;
    ob.b[0] = f2bf(x.x); ob.b[1] = f2bf(x.y); ob.b[2] = f2bf(x.z); ob.b[3] = f2bf(x.w);
    *(unsigned long long*)(outB + o) = ob.u;
}

// ---------------------------------------------------------------- WMMA GEMM
// C[M,N] = A[M,K] @ B + bias, bf16 inputs, f32 accumulate, double-buffered LDS,
// 128x128 block tile, 8 waves in 4(M)x2(N) grid, 32x64 per wave = 8 WMMA/K-step.
// TRANSB==0: B is [K,N] row-major (ldb=N). TRANSB==1: B is [N,K] row-major (ldb=K).
// epi==1: out=(acc+bias+r1)*s ; epi==2: out=((acc+bias+r1)*s + r2)*s ; s=sqrt(0.5)
#define BM 128
#define BN 128
#define BK 32
#define KPAD 40

template <int TRANSB>
__global__ __launch_bounds__(256) void k_gemm(
    const bf16_t* __restrict__ A, const bf16_t* __restrict__ B,
    const float* __restrict__ bias,
    float* __restrict__ outF, bf16_t* __restrict__ outB,
    const float* __restrict__ r1, const float* __restrict__ r2,
    int N, int Kd, int lda, int ldb, int ldc,
    long strideA, long strideB, long strideC,
    int epi)
{
    __shared__ __align__(16) bf16_t sA[2][BM][KPAD];
    __shared__ __align__(16) bf16_t sB[2][BN][KPAD];   // stored [n][k]

    int tid  = threadIdx.x;
    int lane = tid & 31, wid = tid >> 5;
    int wm = wid & 3, wn = wid >> 2;                // 4x2 wave grid
    int hv  = lane >> 4;                            // half of wave
    int l16 = lane & 15;
    int bn0 = blockIdx.x * BN, bm0 = blockIdx.y * BM;

    A += (long)blockIdx.z * strideA;
    B += (long)blockIdx.z * strideB;
    long co = (long)blockIdx.z * strideC;

    v8f z = {0.f, 0.f, 0.f, 0.f, 0.f, 0.f, 0.f, 0.f};
    v8f acc[2][4] = {{z, z, z, z}, {z, z, z, z}};

    // fixed per-thread tile coordinates (A tile 128x32, B tile 32x128; 2 uint4 each)
    const int ar = tid >> 2, ak8 = (tid & 3) * 8;
    const int btn = tid >> 2, btk8 = (tid & 3) * 8;     // transB path
    const int bkk = tid >> 4, bn8 = (tid & 15) * 8;     // normal path

    auto g_load = [&](int k0, uint4& a0, uint4& a1, uint4& b0, uint4& b1) {
        a0 = *(const uint4*)(A + (long)(bm0 + ar) * lda + k0 + ak8);
        a1 = *(const uint4*)(A + (long)(bm0 + 64 + ar) * lda + k0 + ak8);
        if (TRANSB) {
            b0 = *(const uint4*)(B + (long)(bn0 + btn) * ldb + k0 + btk8);
            b1 = *(const uint4*)(B + (long)(bn0 + 64 + btn) * ldb + k0 + btk8);
        } else {
            b0 = *(const uint4*)(B + (long)(k0 + bkk) * ldb + bn0 + bn8);
            b1 = *(const uint4*)(B + (long)(k0 + 16 + bkk) * ldb + bn0 + bn8);
        }
    };
    auto s_store = [&](int buf, uint4 a0, uint4 a1, uint4 b0, uint4 b1) {
        *(uint4*)&sA[buf][ar][ak8]      = a0;
        *(uint4*)&sA[buf][64 + ar][ak8] = a1;
        if (TRANSB) {
            *(uint4*)&sB[buf][btn][btk8]      = b0;
            *(uint4*)&sB[buf][64 + btn][btk8] = b1;
        } else {
            v8bf v0 = __builtin_bit_cast(v8bf, b0);
            v8bf v1 = __builtin_bit_cast(v8bf, b1);
#pragma unroll
            for (int j = 0; j < 8; ++j) {
                sB[buf][bn8 + j][bkk]      = v0[j];
                sB[buf][bn8 + j][16 + bkk] = v1[j];
            }
        }
    };

    int nK = Kd / BK;
    uint4 a0, a1, b0, b1;
    g_load(0, a0, a1, b0, b1);
    s_store(0, a0, a1, b0, b1);
    __syncthreads();

    for (int kt = 0; kt < nK; ++kt) {
        int buf = kt & 1;
        bool more = (kt + 1 < nK);
        if (more) g_load((kt + 1) * BK, a0, a1, b0, b1); // in flight during MMA
        if (kt + 2 < nK)
            __builtin_prefetch(A + (long)(bm0 + ar) * lda + (kt + 2) * BK + ak8, 0, 1);

        v16bf af[2], bfv[4];
#pragma unroll
        for (int im = 0; im < 2; ++im) {
            int m = wm * 32 + im * 16 + l16;
            // A frag 16x32: lanes0-15 K{0..7,16..23}, lanes16-31 K{8..15,24..31}
            v8bf alo = *(const v8bf*)&sA[buf][m][hv * 8];
            v8bf ahi = *(const v8bf*)&sA[buf][m][16 + hv * 8];
            af[im] = __builtin_shufflevector(alo, ahi,
                0, 1, 2, 3, 4, 5, 6, 7, 8, 9, 10, 11, 12, 13, 14, 15);
        }
#pragma unroll
        for (int in = 0; in < 4; ++in) {
            int nn = wn * 64 + in * 16 + l16;
            // B frag 32x16: lanes0-15 K{0..7,16..23}, lanes16-31 K{8..15,24..31}, N=lane&15
            v8bf blo = *(const v8bf*)&sB[buf][nn][hv * 8];
            v8bf bhi = *(const v8bf*)&sB[buf][nn][16 + hv * 8];
            bfv[in] = __builtin_shufflevector(blo, bhi,
                0, 1, 2, 3, 4, 5, 6, 7, 8, 9, 10, 11, 12, 13, 14, 15);
        }
#pragma unroll
        for (int im = 0; im < 2; ++im)
#pragma unroll
            for (int in = 0; in < 4; ++in)
                acc[im][in] = __builtin_amdgcn_wmma_f32_16x16x32_bf16(
                    false, af[im], false, bfv[in], (short)0, acc[im][in], false, false);

        if (more) s_store(buf ^ 1, a0, a1, b0, b1);      // waits loadcnt here
        __syncthreads();
    }

    const float scale = 0.70710678118654752f;
#pragma unroll
    for (int im = 0; im < 2; ++im) {
#pragma unroll
        for (int in = 0; in < 4; ++in) {
            int m0 = bm0 + wm * 32 + im * 16;
            int n0 = bn0 + wn * 64 + in * 16;
            int col = n0 + l16;
            float bv = bias ? bias[col] : 0.f;
#pragma unroll
            for (int r = 0; r < 8; ++r) {
                int row = m0 + r + hv * 8;          // C layout: VGPR r -> M=r(+8)
                long idx = (long)row * ldc + col;
                float v = acc[im][in][r] + bv;
                if (epi == 1)      v = (v + r1[idx]) * scale;
                else if (epi == 2) v = ((v + r1[idx]) * scale + r2[idx]) * scale;
                if (outF) outF[co + idx] = v;
                if (outB) outB[co + idx] = f2bf(v);
            }
        }
    }
}

// ---------------------------------------------------------------- host
extern "C" void kernel_launch(void* const* d_in, const int* in_sizes, int n_in,
                              void* d_out, int out_size, void* d_ws, size_t ws_size,
                              hipStream_t stream) {
    const int Bb = 4, T = 1024, S = 1024, E = 512, H = 1024, V = 16000, Ll = 10;
    const long M = (long)Bb * T;                    // 4096

    const float* target = (const float*)d_in[0];
    const float* encC   = (const float*)d_in[1];
    const float* encD   = (const float*)d_in[2];
    const float* pos    = (const float*)d_in[3];
    const float* W_e2h  = (const float*)d_in[4];
    const float* b_e2h  = (const float*)d_in[5];
    const float* W_h2e  = (const float*)d_in[6];
    const float* b_h2e  = (const float*)d_in[7];
    const float* W_ah2e = (const float*)d_in[8];
    const float* b_ah2e = (const float*)d_in[9];
    const float* W_ae2h = (const float*)d_in[10];
    const float* b_ae2h = (const float*)d_in[11];
    const float* W_out  = (const float*)d_in[12];
    const float* b_out  = (const float*)d_in[13];
    const float* conv_w = (const float*)d_in[14];
    const float* conv_b = (const float*)d_in[15];

    char* ws = (char*)d_ws;
    size_t off = 0;
    auto alloc = [&](size_t bytes) -> void* {
        void* p = ws + off;
        off += (bytes + 255) & ~(size_t)255;
        return p;
    };

    bf16_t* We2hB  = (bf16_t*)alloc((size_t)E * H * 2);
    bf16_t* Wh2eB  = (bf16_t*)alloc((size_t)H * E * 2);
    bf16_t* Wah2eB = (bf16_t*)alloc((size_t)H * E * 2);
    bf16_t* Wae2hB = (bf16_t*)alloc((size_t)E * H * 2);
    bf16_t* WoutB  = (bf16_t*)alloc((size_t)E * V * 2);
    bf16_t* convWB = (bf16_t*)alloc((size_t)Ll * 3 * H * 2 * H * 2);
    bf16_t* encCB  = (bf16_t*)alloc((size_t)Bb * S * E * 2);
    bf16_t* encDB  = (bf16_t*)alloc((size_t)Bb * S * E * 2);
    float*  embF   = (float*) alloc((size_t)M * E * 4);
    bf16_t* embB   = (bf16_t*)alloc((size_t)M * E * 2);
    float*  convInF= (float*) alloc((size_t)M * H * 4);
    bf16_t* im2colB= (bf16_t*)alloc((size_t)M * 3 * H * 2);
    float*  c2hF   = (float*) alloc((size_t)M * 2 * H * 4);
    float*  convedF= (float*) alloc((size_t)M * H * 4);
    bf16_t* convedB= (bf16_t*)alloc((size_t)M * H * 2);
    bf16_t* combB  = (bf16_t*)alloc((size_t)M * E * 2);
    float*  energyF= (float*) alloc((size_t)Bb * T * S * 4);
    bf16_t* attnB  = (bf16_t*)alloc((size_t)Bb * T * S * 2);
    bf16_t* attEB  = (bf16_t*)alloc((size_t)M * E * 2);
    bf16_t* xB     = (bf16_t*)alloc((size_t)M * H * 2);
    bf16_t* ceB    = (bf16_t*)alloc((size_t)M * E * 2);

    float* outV  = (float*)d_out;                   // [B,T,V]
    float* attnF = (float*)d_out + M * (long)V;     // [B,T,S]

    auto cvt = [&](const float* s, bf16_t* d, long n) {
        long nt = (n + 3) / 4;
        k_f32_to_bf16<<<dim3((unsigned)((nt + 255) / 256)), dim3(256), 0, stream>>>(s, d, n);
    };
    auto gemm = [&](const bf16_t* Ap, const bf16_t* Bp, const float* bp,
                    float* oF, bf16_t* oB, const float* r1, const float* r2,
                    int Mm, int Nn, int Kd, int lda, int ldb, int ldc,
                    long sA_, long sB_, long sC_, int nb, int tB, int epi) {
        dim3 g((unsigned)(Nn / BN), (unsigned)(Mm / BM), (unsigned)nb);
        if (tB)
            k_gemm<1><<<g, dim3(256), 0, stream>>>(Ap, Bp, bp, oF, oB, r1, r2,
                                                   Nn, Kd, lda, ldb, ldc,
                                                   sA_, sB_, sC_, epi);
        else
            k_gemm<0><<<g, dim3(256), 0, stream>>>(Ap, Bp, bp, oF, oB, r1, r2,
                                                   Nn, Kd, lda, ldb, ldc,
                                                   sA_, sB_, sC_, epi);
    };

    // one-time converts (redone every call: no cached state)
    cvt(W_e2h,  We2hB,  (long)E * H);
    cvt(W_h2e,  Wh2eB,  (long)H * E);
    cvt(W_ah2e, Wah2eB, (long)H * E);
    cvt(W_ae2h, Wae2hB, (long)E * H);
    cvt(W_out,  WoutB,  (long)E * V);
    cvt(conv_w, convWB, (long)Ll * 3 * H * 2 * H);
    cvt(encC,   encCB,  (long)Bb * S * E);
    cvt(encD,   encDB,  (long)Bb * S * E);

    // embedded = target + pos
    {
        long n = M * E;
        k_embed<<<dim3((unsigned)((n / 4 + 255) / 256)), dim3(256), 0, stream>>>(
            target, pos, embF, embB, (long)T * E, n);
    }
    // conv_input = embedded @ W_e2h + b_e2h
    gemm(embB, We2hB, b_e2h, convInF, nullptr, nullptr, nullptr,
         (int)M, H, E, E, H, H, 0, 0, 0, 1, 0, 0);

    for (int i = 0; i < Ll; ++i) {
        const bf16_t* Wc = convWB + (long)i * 3 * H * 2 * H;
        const float*  bc = conv_b + (long)i * 2 * H;
        long nIm = M * 3L * H;
        k_im2col<<<dim3((unsigned)((nIm / 4 + 255) / 256)), dim3(256), 0, stream>>>(
            convInF, im2colB, T, H, nIm);
        // conved2h = im2col @ conv_w[i] + conv_b[i]
        gemm(im2colB, Wc, bc, c2hF, nullptr, nullptr, nullptr,
             (int)M, 2 * H, 3 * H, 3 * H, 2 * H, 2 * H, 0, 0, 0, 1, 0, 0);
        long nG = M * (long)H;
        k_glu<<<dim3((unsigned)((nG / 4 + 255) / 256)), dim3(256), 0, stream>>>(
            c2hF, convedF, convedB, H, nG);
        // combined = (conved @ W_ah2e + b_ah2e + embedded) * scale   -> bf16
        gemm(convedB, Wah2eB, b_ah2e, nullptr, combB, embF, nullptr,
             (int)M, E, H, H, E, E, 0, 0, 0, 1, 0, 1);
        // energy = combined @ encoder_conved^T   (batched, transB)
        gemm(combB, encCB, nullptr, energyF, nullptr, nullptr, nullptr,
             T, S, E, E, E, S, (long)T * E, (long)S * E, (long)T * S, Bb, 1, 0);
        // attention = softmax(energy)  -> d_out (f32) + bf16 copy
        k_softmax<<<dim3((unsigned)M), dim3(256), 0, stream>>>(energyF, attnF, attnB, S);
        // attended_e = attention @ encoder_combined   (batched)
        gemm(attnB, encDB, nullptr, nullptr, attEB, nullptr, nullptr,
             T, E, S, S, E, E, (long)T * S, (long)S * E, (long)T * E, Bb, 0, 0);
        // conv_input = ((conved + (attended_e @ W_ae2h + b_ae2h)) * s + conv_input) * s
        gemm(attEB, Wae2hB, b_ae2h, convInF, nullptr, convedF, convInF,
             (int)M, H, E, E, H, H, 0, 0, 0, 1, 0, 2);
    }

    // conved = conv_input @ W_h2e + b_h2e ; output = conved @ W_out + b_out
    cvt(convInF, xB, M * (long)H);
    gemm(xB, Wh2eB, b_h2e, nullptr, ceB, nullptr, nullptr,
         (int)M, E, H, H, E, E, 0, 0, 0, 1, 0, 0);
    gemm(ceB, WoutB, b_out, outV, nullptr, nullptr, nullptr,
         (int)M, V, E, E, V, V, 0, 0, 0, 1, 0, 0);

    (void)in_sizes; (void)n_in; (void)out_size; (void)ws_size;
}
